// PointTactileTokenizer_59493886984773
// MI455X (gfx1250) — matrix operand order repaired and checked by reading, not compiled, not run on hardware
//
#include <hip/hip_runtime.h>
#include <hip/hip_bf16.h>
#include <math.h>

// ---------------------------------------------------------------------------
// Problem constants (from reference)
// ---------------------------------------------------------------------------
#define BB        8
#define NN        8192
#define MM        2048
#define DD        256
#define IN_P      138   // point MLP input features
#define IN_P_PAD  160   // padded to multiple of 32 (5 k-tiles)
#define IN_T      52    // tactile MLP input features
#define IN_T_PAD  64    // padded to multiple of 32 (2 k-tiles)
#define TOK_ROWS  (1 + NN + MM)   // 10241
#define KSEL      8
#define INV_TEMP  20.0f           // 1 / 0.05

typedef _Float16 h16v __attribute__((ext_vector_type(16)));
typedef _Float16 h8v  __attribute__((ext_vector_type(8)));
typedef float    f8v  __attribute__((ext_vector_type(8)));

// ---------------------------------------------------------------------------
// 1) Build padded f16 input matrices (feats ++ positional-encoding ++ [tri])
//    PE layout per coord: sin(b0..b5), cos(b0..b5); freqs = linspace(1,10,6)*pi
// ---------------------------------------------------------------------------
__device__ __forceinline__ void pe_write(_Float16* out, float x) {
    const float fr[6] = {1.0f, 2.8f, 4.6f, 6.4f, 8.2f, 10.0f};
#pragma unroll
    for (int b = 0; b < 6; ++b) {
        float ang = x * fr[b] * 3.14159265358979323846f;
        out[b]     = (_Float16)__sinf(ang);
        out[6 + b] = (_Float16)__cosf(ang);
    }
}

__global__ __launch_bounds__(256)
void build_point_inputs(const float* __restrict__ xyz,
                        const float* __restrict__ feats,
                        const float* __restrict__ tri,
                        _Float16* __restrict__ Xp)
{
    int i = blockIdx.x * 256 + threadIdx.x;          // 0 .. B*N-1
    if (i >= BB * NN) return;
    _Float16* out = Xp + (size_t)i * IN_P_PAD;
    const float* f = feats + (size_t)i * 6;
#pragma unroll
    for (int j = 0; j < 6; ++j) out[j] = (_Float16)f[j];
    const float* p = xyz + (size_t)i * 3;
#pragma unroll
    for (int c = 0; c < 3; ++c) pe_write(out + 6 + c * 12, p[c]);
    const float* tr = tri + (size_t)i * 96;
#pragma unroll 8
    for (int j = 0; j < 96; ++j) out[42 + j] = (_Float16)tr[j];
#pragma unroll
    for (int j = IN_P; j < IN_P_PAD; ++j) out[j] = (_Float16)0.0f;
}

__global__ __launch_bounds__(256)
void build_tactile_inputs(const float* __restrict__ xyz,
                          const float* __restrict__ feats,
                          _Float16* __restrict__ Xt)
{
    int i = blockIdx.x * 256 + threadIdx.x;          // 0 .. B*M-1
    if (i >= BB * MM) return;
    _Float16* out = Xt + (size_t)i * IN_T_PAD;
    const float* f = feats + (size_t)i * 16;
#pragma unroll
    for (int j = 0; j < 16; ++j) out[j] = (_Float16)f[j];
    const float* p = xyz + (size_t)i * 3;
#pragma unroll
    for (int c = 0; c < 3; ++c) pe_write(out + 16 + c * 12, p[c]);
#pragma unroll
    for (int j = IN_T; j < IN_T_PAD; ++j) out[j] = (_Float16)0.0f;
}

// ---------------------------------------------------------------------------
// 2) Pack f32 weight [K][256] -> f16 transposed [256][KPAD] (zero padded)
// ---------------------------------------------------------------------------
__global__ __launch_bounds__(256)
void pack_weight_t(const float* __restrict__ W, _Float16* __restrict__ WT,
                   int K, int KPAD)
{
    int o = blockIdx.x * 256 + threadIdx.x;
    if (o >= DD * KPAD) return;
    int n  = o / KPAD;
    int kp = o - n * KPAD;
    WT[o] = (kp < K) ? (_Float16)W[(size_t)kp * DD + n] : (_Float16)0.0f;
}

// ---------------------------------------------------------------------------
// 3) WMMA GEMM layer:  Y = act(X[rows x KPAD] @ WT^T + bias)
//    block = 128 threads = 4 waves; wave owns a 32(M) x 64(N) tile
//    (two 16-row A fragments share each B fragment -> 2 WMMAs per B load).
//    A frag (16x32 f16): lane L (r=L&15,h=L>>4) holds K runs [h*8,+8) and
//    [16+h*8,+8) of row r  -> two contiguous b128 loads.
//    B frag (32x16 f16): lane L holds col n=L&15, K runs [(L>>4)*8,+8)
//    and [16+(L>>4)*8,+8) of WT row n -> two contiguous b128 loads.
//    FINAL=false: GELU(erf) + f16 store; FINAL=true: plain f32 store.
// ---------------------------------------------------------------------------
template <int KPAD, bool FINAL>
__global__ __launch_bounds__(128)
void wmma_mlp_layer(const _Float16* __restrict__ X,
                    const _Float16* __restrict__ WT,
                    const float* __restrict__ bias,
                    void* __restrict__ Yv)
{
    const int lane  = threadIdx.x & 31;
    const int wave  = threadIdx.x >> 5;
    const int mbase = blockIdx.x * 128 + wave * 32;   // 32 M-rows per wave
    const int nbase = blockIdx.y * 64;
    const int r15   = lane & 15;
    const int hsel  = lane >> 4;

    f8v acc0[4], acc1[4];
#pragma unroll
    for (int t = 0; t < 4; ++t)
#pragma unroll
        for (int r = 0; r < 8; ++r) { acc0[t][r] = 0.0f; acc1[t][r] = 0.0f; }

    const _Float16* xrow0 = X + (size_t)(mbase + r15) * KPAD;
    const _Float16* xrow1 = X + (size_t)(mbase + 16 + r15) * KPAD;

#pragma unroll 2
    for (int kt = 0; kt < KPAD; kt += 32) {
        // ---- two A fragments: 4x b128 loads ----
        h16v a0, a1;
        {
            const h8v l0 = *(const h8v*)(xrow0 + kt + hsel * 8);
            const h8v h0 = *(const h8v*)(xrow0 + kt + 16 + hsel * 8);
            const h8v l1 = *(const h8v*)(xrow1 + kt + hsel * 8);
            const h8v h1 = *(const h8v*)(xrow1 + kt + 16 + hsel * 8);
#pragma unroll
            for (int j = 0; j < 8; ++j) {
                a0[j] = l0[j]; a0[8 + j] = h0[j];
                a1[j] = l1[j]; a1[8 + j] = h1[j];
            }
        }
        // ---- 4 N-tiles: load B fragment once, feed both A fragments ----
#pragma unroll
        for (int t = 0; t < 4; ++t) {
            const int n = nbase + t * 16 + r15;
            const _Float16* wrow = WT + (size_t)n * KPAD + kt + hsel * 8;
            h16v bfr;
            {
                const h8v b0 = *(const h8v*)(wrow);
                const h8v b1 = *(const h8v*)(wrow + 16);
#pragma unroll
                for (int j = 0; j < 8; ++j) { bfr[j] = b0[j]; bfr[8 + j] = b1[j]; }
            }
            acc0[t] = __builtin_amdgcn_wmma_f32_16x16x32_f16(
                false, a0, false, bfr, (short)0, acc0[t], false, false);
            acc1[t] = __builtin_amdgcn_wmma_f32_16x16x32_f16(
                false, a1, false, bfr, (short)0, acc1[t], false, false);
        }
    }

    // ---- epilogue: bias (+ exact-erf GELU) ----
#pragma unroll
    for (int t = 0; t < 4; ++t) {
        const int n  = nbase + t * 16 + r15;
        const float bn = bias[n];
#pragma unroll
        for (int r = 0; r < 8; ++r) {
            const int m0 = mbase + hsel * 8 + r;
            const int m1 = m0 + 16;
            float v0 = acc0[t][r] + bn;
            float v1 = acc1[t][r] + bn;
            if (!FINAL) {
                v0 = 0.5f * v0 * (1.0f + erff(v0 * 0.70710678118654752f));
                v1 = 0.5f * v1 * (1.0f + erff(v1 * 0.70710678118654752f));
                ((_Float16*)Yv)[(size_t)m0 * DD + n] = (_Float16)v0;
                ((_Float16*)Yv)[(size_t)m1 * DD + n] = (_Float16)v1;
            } else {
                ((float*)Yv)[(size_t)m0 * DD + n] = v0;
                ((float*)Yv)[(size_t)m1 * DD + n] = v1;
            }
        }
    }
}

// ---------------------------------------------------------------------------
// 4) kNN (k=8) + softmax registration + final point-row assembly.
//    256 threads = 8 waves; one point per wave; tactile xyz staged in LDS.
//    Per-lane sorted top-8 over 64 strided candidates, then 8 rounds of
//    wave-wide argmin extraction via __shfl_xor (wave32).
// ---------------------------------------------------------------------------
__global__ __launch_bounds__(256)
void knn_register_assemble(const float* __restrict__ pxyz,
                           const float* __restrict__ txyz,
                           const float* __restrict__ ptok,
                           const float* __restrict__ ttok,
                           const float* __restrict__ ctx,
                           float* __restrict__ tokens)
{
    __shared__ float st[MM * 3];                     // 24 KB
    const int g0 = blockIdx.x * 8;                   // first global point index
    const int b  = g0 >> 13;                         // N = 8192 = 2^13

    const float* tb = txyz + (size_t)b * MM * 3;
    for (int j = threadIdx.x; j < MM * 3; j += 256) st[j] = tb[j];
    __syncthreads();

    const int lane = threadIdx.x & 31;
    const int wid  = threadIdx.x >> 5;
    const int g    = g0 + wid;                       // this wave's point
    const float px = pxyz[(size_t)g * 3 + 0];
    const float py = pxyz[(size_t)g * 3 + 1];
    const float pz = pxyz[(size_t)g * 3 + 2];

    float d[KSEL];
    int   id[KSEL];
#pragma unroll
    for (int q = 0; q < KSEL; ++q) { d[q] = 3.0e38f; id[q] = 0; }

    // per-lane scan: 2048/32 = 64 candidates, keep sorted top-8
    for (int t = lane; t < MM; t += 32) {
        float dx = st[t * 3 + 0] - px;
        float dy = st[t * 3 + 1] - py;
        float dz = st[t * 3 + 2] - pz;
        float dist = sqrtf(fmaxf(dx * dx + dy * dy + dz * dz, 0.0f));
        if (dist < d[KSEL - 1]) {
            d[KSEL - 1] = dist; id[KSEL - 1] = t;
#pragma unroll
            for (int q = KSEL - 1; q > 0; --q) {
                if (d[q] < d[q - 1]) {
                    float td = d[q];  d[q]  = d[q - 1];  d[q - 1]  = td;
                    int   ti = id[q]; id[q] = id[q - 1]; id[q - 1] = ti;
                }
            }
        }
    }

    // 8 rounds of wave argmin extraction
    float kd[KSEL];
    int   ki[KSEL];
#pragma unroll
    for (int r = 0; r < KSEL; ++r) {
        float ld = d[0]; int lq = 0;
#pragma unroll
        for (int q = 1; q < KSEL; ++q) if (d[q] < ld) { ld = d[q]; lq = q; }
        int payload = (((lane << 3) | lq) << 12) | id[lq];   // slot | idx
        float bd = ld; int bp = payload;
#pragma unroll
        for (int off = 16; off > 0; off >>= 1) {
            float od = __shfl_xor(bd, off, 32);
            int   op = __shfl_xor(bp, off, 32);
            if (od < bd || (od == bd && op < bp)) { bd = od; bp = op; }
        }
        kd[r] = bd;
        ki[r] = bp & 4095;
        int slot = bp >> 12;
        if ((slot >> 3) == lane) d[slot & 7] = 3.0e38f;     // remove winner
    }

    // softmax over -dist/temp (kd[0] is the minimum)
    float w[KSEL], s = 0.0f;
#pragma unroll
    for (int r = 0; r < KSEL; ++r) { w[r] = expf(-(kd[r] - kd[0]) * INV_TEMP); s += w[r]; }
    const float inv = 1.0f / s;
#pragma unroll
    for (int r = 0; r < KSEL; ++r) w[r] *= inv;

    // weighted gather + point token + ctx -> output point row
    const int c0 = lane * 8;                         // 8 channels per lane
    float acc[8];
    const float* pr = ptok + (size_t)g * DD + c0;
#pragma unroll
    for (int r = 0; r < 8; ++r) acc[r] = pr[r];
#pragma unroll
    for (int q = 0; q < KSEL; ++q) {
        const float* tr = ttok + ((size_t)b * MM + ki[q]) * DD + c0;
        const float wq = w[q];
#pragma unroll
        for (int r = 0; r < 8; ++r) acc[r] = fmaf(wq, tr[r], acc[r]);
    }
    const float* cr = ctx + (size_t)b * DD + c0;
    const int nloc = g & (NN - 1);
    float* orow = tokens + ((size_t)b * TOK_ROWS + 1 + nloc) * DD + c0;
#pragma unroll
    for (int r = 0; r < 8; ++r) orow[r] = acc[r] + cr[r];
}

// ---------------------------------------------------------------------------
// 5) Remaining output rows
// ---------------------------------------------------------------------------
__global__ __launch_bounds__(256)
void assemble_global_rows(const float* __restrict__ gtok,
                          const float* __restrict__ ctx,
                          float* __restrict__ tokens)
{
    int i = blockIdx.x * 256 + threadIdx.x;          // B*D
    if (i >= BB * DD) return;
    int b = i >> 8, dch = i & 255;
    tokens[((size_t)b * TOK_ROWS) * DD + dch] = gtok[dch] + ctx[(size_t)b * DD + dch];
}

__global__ __launch_bounds__(256)
void assemble_tactile_rows(const float* __restrict__ ttok,
                           const float* __restrict__ ctx,
                           float* __restrict__ tokens)
{
    int i = blockIdx.x * 256 + threadIdx.x;          // B*M*D
    if (i >= BB * MM * DD) return;
    int dch = i & 255;
    int m   = (i >> 8) & (MM - 1);
    int b   = i >> 19;                               // / (2048*256)
    tokens[((size_t)b * TOK_ROWS + 1 + NN + m) * DD + dch] =
        ttok[(size_t)i] + ctx[(size_t)b * DD + dch];
}

// ---------------------------------------------------------------------------
// Host launcher
// ---------------------------------------------------------------------------
extern "C" void kernel_launch(void* const* d_in, const int* in_sizes, int n_in,
                              void* d_out, int out_size, void* d_ws, size_t ws_size,
                              hipStream_t stream)
{
    (void)in_sizes; (void)n_in; (void)out_size; (void)ws_size;

    const float* pxyz = (const float*)d_in[0];
    const float* pfeat= (const float*)d_in[1];
    const float* txyz = (const float*)d_in[2];
    const float* tfeat= (const float*)d_in[3];
    const float* tri  = (const float*)d_in[4];
    const float* ctx  = (const float*)d_in[5];
    const float* gtok = (const float*)d_in[6];
    const float* Wp1  = (const float*)d_in[7];  const float* bp1 = (const float*)d_in[8];
    const float* Wp2  = (const float*)d_in[9];  const float* bp2 = (const float*)d_in[10];
    const float* Wp3  = (const float*)d_in[11]; const float* bp3 = (const float*)d_in[12];
    const float* Wt1  = (const float*)d_in[13]; const float* bt1 = (const float*)d_in[14];
    const float* Wt2  = (const float*)d_in[15]; const float* bt2 = (const float*)d_in[16];
    const float* Wt3  = (const float*)d_in[17]; const float* bt3 = (const float*)d_in[18];
    float* tokens = (float*)d_out;

    // workspace carve-out (~192 MB total)
    char*  base = (char*)d_ws;
    size_t off  = 0;
    auto carve = [&](size_t bytes) -> void* {
        void* p = base + off;
        off += (bytes + 255) & ~(size_t)255;
        return p;
    };
    _Float16* Xp   = (_Float16*)carve((size_t)BB * NN * IN_P_PAD * 2);
    _Float16* Xt   = (_Float16*)carve((size_t)BB * MM * IN_T_PAD * 2);
    _Float16* WpT1 = (_Float16*)carve((size_t)DD * IN_P_PAD * 2);
    _Float16* WpT2 = (_Float16*)carve((size_t)DD * DD * 2);
    _Float16* WpT3 = (_Float16*)carve((size_t)DD * DD * 2);
    _Float16* WtT1 = (_Float16*)carve((size_t)DD * IN_T_PAD * 2);
    _Float16* WtT2 = (_Float16*)carve((size_t)DD * DD * 2);
    _Float16* WtT3 = (_Float16*)carve((size_t)DD * DD * 2);
    _Float16* Hp1  = (_Float16*)carve((size_t)BB * NN * DD * 2);
    _Float16* Hp2  = (_Float16*)carve((size_t)BB * NN * DD * 2);
    _Float16* Ht1  = (_Float16*)carve((size_t)BB * MM * DD * 2);
    _Float16* Ht2  = (_Float16*)carve((size_t)BB * MM * DD * 2);
    float*    ptok = (float*)carve((size_t)BB * NN * DD * 4);
    float*    ttok = (float*)carve((size_t)BB * MM * DD * 4);

    // 1) build padded f16 inputs
    build_point_inputs<<<(BB * NN + 255) / 256, 256, 0, stream>>>(pxyz, pfeat, tri, Xp);
    build_tactile_inputs<<<(BB * MM + 255) / 256, 256, 0, stream>>>(txyz, tfeat, Xt);

    // 2) pack weights (transpose + f16)
    pack_weight_t<<<(DD * IN_P_PAD + 255) / 256, 256, 0, stream>>>(Wp1, WpT1, IN_P, IN_P_PAD);
    pack_weight_t<<<(DD * DD + 255) / 256, 256, 0, stream>>>(Wp2, WpT2, DD, DD);
    pack_weight_t<<<(DD * DD + 255) / 256, 256, 0, stream>>>(Wp3, WpT3, DD, DD);
    pack_weight_t<<<(DD * IN_T_PAD + 255) / 256, 256, 0, stream>>>(Wt1, WtT1, IN_T, IN_T_PAD);
    pack_weight_t<<<(DD * DD + 255) / 256, 256, 0, stream>>>(Wt2, WtT2, DD, DD);
    pack_weight_t<<<(DD * DD + 255) / 256, 256, 0, stream>>>(Wt3, WtT3, DD, DD);

    // 3) point MLP: 65536 rows, 128-row block tile -> grid (512, 4)
    dim3 gp((BB * NN) / 128, DD / 64);
    wmma_mlp_layer<IN_P_PAD, false><<<gp, 128, 0, stream>>>(Xp,  WpT1, bp1, Hp1);
    wmma_mlp_layer<DD,       false><<<gp, 128, 0, stream>>>(Hp1, WpT2, bp2, Hp2);
    wmma_mlp_layer<DD,       true ><<<gp, 128, 0, stream>>>(Hp2, WpT3, bp3, ptok);

    // tactile MLP: 16384 rows -> grid (128, 4)
    dim3 gt((BB * MM) / 128, DD / 64);
    wmma_mlp_layer<IN_T_PAD, false><<<gt, 128, 0, stream>>>(Xt,  WtT1, bt1, Ht1);
    wmma_mlp_layer<DD,       false><<<gt, 128, 0, stream>>>(Ht1, WtT2, bt2, Ht2);
    wmma_mlp_layer<DD,       true ><<<gt, 128, 0, stream>>>(Ht2, WtT3, bt3, ttok);

    // 4) kNN + registration + point rows (one wave per point, 8 waves/block)
    knn_register_assemble<<<(BB * NN) / 8, 256, 0, stream>>>(
        pxyz, txyz, ptok, ttok, ctx, tokens);

    // 5) global + tactile rows
    assemble_global_rows<<<(BB * DD + 255) / 256, 256, 0, stream>>>(gtok, ctx, tokens);
    assemble_tactile_rows<<<(BB * MM * DD + 255) / 256, 256, 0, stream>>>(ttok, ctx, tokens);
}
